// MultiHeadAttention_67542655697553
// MI455X (gfx1250) — matrix operand order repaired and checked by reading, not compiled, not run on hardware
//
#include <hip/hip_runtime.h>
#include <cmath>

#define DM    1024
#define SLEN  2048
#define BATCH 2
#define NH    16
#define DEPTH 64

typedef __attribute__((ext_vector_type(16))) _Float16 v16h;
typedef __attribute__((ext_vector_type(8)))  _Float16 v8h;
typedef __attribute__((ext_vector_type(4)))  _Float16 v4h;
typedef __attribute__((ext_vector_type(8)))  float    v8f;
typedef __attribute__((ext_vector_type(4)))  unsigned int u32x4;
typedef __attribute__((ext_vector_type(4)))  int          i32x4;
typedef __attribute__((ext_vector_type(8)))  int          i32x8;

static __device__ __forceinline__ int lane_id() { return threadIdx.x & 31; }

static __device__ __forceinline__ v8f wmma_f16(v16h a, v16h b, v8f c) {
  // emits v_wmma_f32_16x16x32_f16
  return __builtin_amdgcn_wmma_f32_16x16x32_f16(false, a, false, b, (short)0, c,
                                                false, false);
}

// A-fragment: 16x32 (MxK) f16 tile from row-major memory, row stride ldk.
// Contiguous k order (same as B): lane L<16 -> row M=L, K=0..15;
//             lane L+16 -> row M=L, K=16..31.
static __device__ __forceinline__ v16h frag_a(const _Float16* base, int ldk) {
  int lane = lane_id();
  int m  = lane & 15;
  int k0 = (lane & 16) ? 16 : 0;
  const _Float16* p = base + (size_t)m * ldk + k0;
  v8h lo = *(const v8h*)(p);
  v8h hi = *(const v8h*)(p + 8);
  v16h f;
#pragma unroll
  for (int i = 0; i < 8; ++i) { f[i] = lo[i]; f[8 + i] = hi[i]; }
  return f;
}

// B-fragment: 32x16 (KxN) f16 tile from the N-major view Bt[n][k], row stride
// ldk. Contiguous k order (same as A): lane L<16 -> col N=L holds K=0..15; lane L+16 -> K=16..31.
static __device__ __forceinline__ v16h frag_b(const _Float16* baseNk, int ldk) {
  int lane = lane_id();
  int n  = lane & 15;
  int k0 = (lane & 16) ? 16 : 0;
  const _Float16* p = baseNk + (size_t)n * ldk + k0;
  v8h lo = *(const v8h*)(p);
  v8h hi = *(const v8h*)(p + 8);
  v16h f;
#pragma unroll
  for (int i = 0; i < 8; ++i) { f[i] = lo[i]; f[8 + i] = hi[i]; }
  return f;
}

// B-fragment from the TDM-staged LDS panel.
// Panel layout (set by D# pad fields): row n at byte n*2080; element k at
// +2k (+16 extra once k >= 512 because of the mid-row 16B pad).
#define BPANEL_ROW 2080
static __device__ __forceinline__ v16h frag_b_panel(const char* Bl, int colBase,
                                                    int kk) {
  int lane = lane_id();
  int n = colBase + (lane & 15);
  int k = kk + ((lane & 16) ? 16 : 0);
  const char* p = Bl + n * BPANEL_ROW + 2 * k + ((k >= 512) ? 16 : 0);
  v8h lo = *(const v8h*)(p);
  v8h hi = *(const v8h*)(p + 16);
  v16h f;
#pragma unroll
  for (int i = 0; i < 8; ++i) { f[i] = lo[i]; f[8 + i] = hi[i]; }
  return f;
}

// Issue a TDM load of a 64-row x 1024-col f16 panel (row stride 1024) from
// global memory into LDS, with 16B padding every 1024B for bank decorrelation.
static __device__ __forceinline__ void tdm_load_b_panel(const _Float16* gsrc,
                                                        unsigned lds_addr) {
  unsigned long long ga = (unsigned long long)(size_t)gsrc;
  u32x4 g0;
  g0[0] = 1u;                                   // count=1, user descriptor
  g0[1] = lds_addr;                             // LDS byte address
  g0[2] = (unsigned)(ga & 0xffffffffu);         // global_addr[31:0]
  g0[3] = (unsigned)((ga >> 32) & 0x01ffffffu)  // global_addr[56:32]
          | (2u << 30);                         // type=2 ("image")
  i32x8 g1;
  g1[0] = 0x07D10000;          // data_size=2B | pad_enable | interval=256dw | amount=4dw
  g1[1] = (int)(1024u << 16);  // tensor_dim0 = 1024 (bits [63:48])
  g1[2] = (int)(1024u << 16);  // tensor_dim1 = 1024 (bits [95:80])
  g1[3] = (int)(1024u << 16);  // tile_dim0   = 1024 (bits [127:112])
  g1[4] = 64;                  // tile_dim1   = 64
  g1[5] = 1024;                // tensor_dim0_stride = 1024 elements
  g1[6] = 0;
  g1[7] = 0;
  i32x4 z4 = {0, 0, 0, 0};
#if defined(__clang_major__) && (__clang_major__ >= 23)
  i32x8 z8 = {0, 0, 0, 0, 0, 0, 0, 0};
  __builtin_amdgcn_tensor_load_to_lds(g0, g1, z4, z4, z8, 0);
#else
  __builtin_amdgcn_tensor_load_to_lds(g0, g1, z4, z4, 0);
#endif
}

// -------------------- LayerNorm + cast to f16 --------------------
__global__ void __launch_bounds__(256)
ln_cast(const float* __restrict__ q, const float* __restrict__ k,
        const float* __restrict__ v, const float* __restrict__ g,
        const float* __restrict__ beta,
        _Float16* __restrict__ oq, _Float16* __restrict__ ok,
        _Float16* __restrict__ ov) {
  __shared__ float sm[18];
  int row = blockIdx.x;
  int which = blockIdx.y;
  const float* x = (which == 0 ? q : which == 1 ? k : v) + (size_t)row * DM;
  _Float16*    o = (which == 0 ? oq : which == 1 ? ok : ov) + (size_t)row * DM;
  int tid = threadIdx.x;
  float4 xv = ((const float4*)x)[tid];
  float s  = xv.x + xv.y + xv.z + xv.w;
  float s2 = xv.x * xv.x + xv.y * xv.y + xv.z * xv.z + xv.w * xv.w;
#pragma unroll
  for (int off = 16; off > 0; off >>= 1) {
    s  += __shfl_xor(s,  off, 32);
    s2 += __shfl_xor(s2, off, 32);
  }
  if ((tid & 31) == 0) { sm[tid >> 5] = s; sm[8 + (tid >> 5)] = s2; }
  __syncthreads();
  if (tid == 0) {
    float a = 0.f, b2 = 0.f;
#pragma unroll
    for (int i = 0; i < 8; ++i) { a += sm[i]; b2 += sm[8 + i]; }
    sm[16] = a; sm[17] = b2;
  }
  __syncthreads();
  float mean = sm[16] * (1.0f / DM);
  float var  = sm[17] * (1.0f / DM) - mean * mean;
  float rstd = rsqrtf(var + 1e-5f);
  float4 gv = ((const float4*)g)[tid];
  float4 bv = ((const float4*)beta)[tid];
  v4h ov4 = { (_Float16)((xv.x - mean) * rstd * gv.x + bv.x),
              (_Float16)((xv.y - mean) * rstd * gv.y + bv.y),
              (_Float16)((xv.z - mean) * rstd * gv.z + bv.z),
              (_Float16)((xv.w - mean) * rstd * gv.w + bv.w) };
  *(v4h*)(o + (size_t)tid * 4) = ov4;
}

// -------------------- weight cast f32 -> f16 --------------------
__global__ void __launch_bounds__(256)
cast_w(const float* __restrict__ w0, const float* __restrict__ w1,
       const float* __restrict__ w2, const float* __restrict__ w3,
       _Float16* __restrict__ o0, _Float16* __restrict__ o1,
       _Float16* __restrict__ o2, _Float16* __restrict__ o3) {
  int which = blockIdx.y;
  const float* w = which == 0 ? w0 : which == 1 ? w1 : which == 2 ? w2 : w3;
  _Float16*    o = which == 0 ? o0 : which == 1 ? o1 : which == 2 ? o2 : o3;
  size_t i = ((size_t)blockIdx.x * 256 + threadIdx.x) * 4;
  float4 val = *(const float4*)(w + i);
  v4h ov = { (_Float16)val.x, (_Float16)val.y, (_Float16)val.z, (_Float16)val.w };
  *(v4h*)(o + i) = ov;
}

// -------------------- WMMA GEMM: Y = A @ Bt^T + bias --------------------
// A: [M, 1024] f16 row-major.  Bt: [1024(N), 1024(K)] f16 row-major (== W).
// Block = 8 waves; block tile = 256 rows x 64 cols; wave tile = 32x64.
// The 64x1024 B panel is staged once per block into LDS by the TDM.
// mode 0: f16 out [M, N] row-major      (Q, K projections)
// mode 1: f16 out transposed [B,H,D,S]  (V projection)
// mode 2: f32 out [M, N] row-major      (final projection -> d_out)
__global__ void __launch_bounds__(256)
gemm16(const _Float16* __restrict__ A, const _Float16* __restrict__ Bt,
       const float* __restrict__ bias, void* __restrict__ outp,
       int M, int mode) {
  extern __shared__ char Bl[];  // 64 * 2080 = 133120 bytes
  int colGrp   = blockIdx.x & 15;
  int rowBlock = blockIdx.x >> 4;
  int wv       = threadIdx.x >> 5;

  if (wv == 0) {
    tdm_load_b_panel(Bt + (size_t)colGrp * 64 * DM,
                     (unsigned)(size_t)(void*)Bl);
    __builtin_amdgcn_s_wait_tensorcnt(0);
  }
  __syncthreads();

  int rowTile = rowBlock * 256 + wv * 32;   // 32 rows per wave
  if (rowTile >= M) return;
  const _Float16* Abase = A + (size_t)rowTile * DM;

  v8f acc[2][4] = {};
  for (int kk = 0; kk < DM; kk += 32) {
    __builtin_prefetch(Abase + kk + 256, 0, 3);
    v16h a0 = frag_a(Abase + kk, DM);
    v16h a1 = frag_a(Abase + (size_t)16 * DM + kk, DM);
#pragma unroll
    for (int t = 0; t < 4; ++t) {
      v16h b = frag_b_panel(Bl, t * 16, kk);
      acc[0][t] = wmma_f16(a0, b, acc[0][t]);
      acc[1][t] = wmma_f16(a1, b, acc[1][t]);
    }
  }

  int lane = lane_id();
  int nlo  = lane & 15;
  int half = (lane >> 4) & 1;
#pragma unroll
  for (int rs = 0; rs < 2; ++rs) {
#pragma unroll
    for (int t = 0; t < 4; ++t) {
      int n = colGrp * 64 + t * 16 + nlo;
      float bv = bias[n];
#pragma unroll
      for (int r = 0; r < 8; ++r) {
        int m = rowTile + rs * 16 + r + half * 8;
        float val = acc[rs][t][r] + bv;
        if (mode == 0) {
          ((_Float16*)outp)[(size_t)m * DM + n] = (_Float16)val;
        } else if (mode == 1) {
          int bb = m / SLEN;
          int ss = m - bb * SLEN;
          ((_Float16*)outp)[((size_t)bb * DM + n) * SLEN + ss] = (_Float16)val;
        } else {
          ((float*)outp)[(size_t)m * DM + n] = val;
        }
      }
    }
  }
}

// -------------------- flash attention (1 wave per 16-query tile) -------------
// Qp, Kp: [B, S, H*D] f16.  Vt: [B, H, D, S] f16.  Ao: [B, S, H*D] f16.
__global__ void __launch_bounds__(32)
attn16(const _Float16* __restrict__ Qp, const _Float16* __restrict__ Kp,
       const _Float16* __restrict__ Vt, _Float16* __restrict__ Ao) {
  __shared__ _Float16 Pl[16 * 32];  // P tile staged for C-layout -> A-layout
  int blk = blockIdx.x;
  int qt = blk & (SLEN / 16 - 1);
  int h  = (blk >> 7) & (NH - 1);
  int b  = blk >> 11;
  int lane = lane_id();
  int nlo  = lane & 15;
  int half = (lane >> 4) & 1;

  const _Float16* Qb = Qp + ((size_t)(b * SLEN + qt * 16)) * DM + h * DEPTH;
  v16h aQ0 = frag_a(Qb, DM);
  v16h aQ1 = frag_a(Qb + 32, DM);

  const _Float16* Kb = Kp + ((size_t)b * SLEN) * DM + h * DEPTH;
  const _Float16* Vb = Vt + ((size_t)(b * NH + h)) * DEPTH * SLEN;

  float mrow[8], lrow[8];
  v8f acc[4] = {};
#pragma unroll
  for (int r = 0; r < 8; ++r) { mrow[r] = -__builtin_inff(); lrow[r] = 0.f; }
  const float scale = 0.125f;  // 1/sqrt(64)

  for (int j = 0; j < SLEN; j += 32) {
    v8f c0 = {}, c1 = {};
    {
      v16h b00 = frag_b(Kb + (size_t)j * DM, DM);
      v16h b01 = frag_b(Kb + (size_t)j * DM + 32, DM);
      c0 = wmma_f16(aQ0, b00, c0);
      c0 = wmma_f16(aQ1, b01, c0);
      v16h b10 = frag_b(Kb + (size_t)(j + 16) * DM, DM);
      v16h b11 = frag_b(Kb + (size_t)(j + 16) * DM + 32, DM);
      c1 = wmma_f16(aQ0, b10, c1);
      c1 = wmma_f16(aQ1, b11, c1);
    }
    float alpha[8], p0[8], p1[8];
#pragma unroll
    for (int r = 0; r < 8; ++r) {
      float s0 = c0[r] * scale;
      float s1 = c1[r] * scale;
      float t = fmaxf(s0, s1);
#pragma unroll
      for (int off = 8; off > 0; off >>= 1) t = fmaxf(t, __shfl_xor(t, off, 16));
      float mn = fmaxf(mrow[r], t);
      float e0 = __expf(s0 - mn);
      float e1 = __expf(s1 - mn);
      float sum = e0 + e1;
#pragma unroll
      for (int off = 8; off > 0; off >>= 1) sum += __shfl_xor(sum, off, 16);
      alpha[r] = __expf(mrow[r] - mn);
      mrow[r] = mn;
      lrow[r] = lrow[r] * alpha[r] + sum;
      p0[r] = e0;
      p1[r] = e1;
    }
    __syncthreads();
#pragma unroll
    for (int r = 0; r < 8; ++r) {
      int m = r + half * 8;
      Pl[m * 32 + nlo]      = (_Float16)p0[r];
      Pl[m * 32 + 16 + nlo] = (_Float16)p1[r];
    }
    __syncthreads();
    v16h aP = frag_a(Pl, 32);
#pragma unroll
    for (int t2 = 0; t2 < 4; ++t2) {
#pragma unroll
      for (int r = 0; r < 8; ++r) acc[t2][r] *= alpha[r];
      v16h bV = frag_b(Vb + (size_t)(t2 * 16) * SLEN + j, SLEN);
      acc[t2] = wmma_f16(aP, bV, acc[t2]);
    }
  }
#pragma unroll
  for (int t2 = 0; t2 < 4; ++t2) {
#pragma unroll
    for (int r = 0; r < 8; ++r) {
      int m = qt * 16 + r + half * 8;
      int d = t2 * 16 + nlo;
      float val = acc[t2][r] / lrow[r];
      Ao[((size_t)(b * SLEN + m)) * DM + h * DEPTH + d] = (_Float16)val;
    }
  }
}

// -------------------- host launch --------------------
extern "C" void kernel_launch(void* const* d_in, const int* in_sizes, int n_in,
                              void* d_out, int out_size, void* d_ws, size_t ws_size,
                              hipStream_t stream) {
  const float* q    = (const float*)d_in[0];
  const float* k    = (const float*)d_in[1];
  const float* v    = (const float*)d_in[2];
  const float* ln_g = (const float*)d_in[3];
  const float* ln_b = (const float*)d_in[4];
  const float* wq_w = (const float*)d_in[5];
  const float* wq_b = (const float*)d_in[6];
  const float* wk_w = (const float*)d_in[7];
  const float* wk_b = (const float*)d_in[8];
  const float* wv_w = (const float*)d_in[9];
  const float* wv_b = (const float*)d_in[10];
  const float* fo_w = (const float*)d_in[11];
  const float* fo_b = (const float*)d_in[12];

  char* ws = (char*)d_ws;
  const size_t MB = 1024 * 1024;
  _Float16* xq   = (_Float16*)(ws + 0 * MB);   // LN(q) f16   [4096,1024]
  _Float16* xk   = (_Float16*)(ws + 8 * MB);
  _Float16* xv   = (_Float16*)(ws + 16 * MB);
  _Float16* wq16 = (_Float16*)(ws + 24 * MB);  // weights f16 [1024,1024]
  _Float16* wk16 = (_Float16*)(ws + 26 * MB);
  _Float16* wv16 = (_Float16*)(ws + 28 * MB);
  _Float16* fo16 = (_Float16*)(ws + 30 * MB);
  _Float16* qp   = (_Float16*)(ws + 32 * MB);  // Q proj [B,S,H*D]
  _Float16* kp   = (_Float16*)(ws + 40 * MB);  // K proj [B,S,H*D]
  _Float16* vt   = (_Float16*)(ws + 48 * MB);  // V proj transposed [B,H,D,S]
  _Float16* ao   = (_Float16*)(ws + 56 * MB);  // attention out [B,S,H*D]

  const int M = BATCH * SLEN;  // 4096

  ln_cast<<<dim3(M, 3), 256, 0, stream>>>(q, k, v, ln_g, ln_b, xq, xk, xv);
  cast_w<<<dim3(DM * DM / (256 * 4), 4), 256, 0, stream>>>(
      wq_w, wk_w, wv_w, fo_w, wq16, wk16, wv16, fo16);

  const int gemmBlocks = (M / 256) * (DM / 64);  // 256 blocks, 8 waves each
  const size_t panelBytes = 64 * BPANEL_ROW;     // 133120 B of LDS
  gemm16<<<gemmBlocks, 256, panelBytes, stream>>>(xq, wq16, wq_b, qp, M, 0);
  gemm16<<<gemmBlocks, 256, panelBytes, stream>>>(xk, wk16, wk_b, kp, M, 0);
  gemm16<<<gemmBlocks, 256, panelBytes, stream>>>(xv, wv16, wv_b, vt, M, 1);

  attn16<<<BATCH * NH * (SLEN / 16), 32, 0, stream>>>(qp, kp, vt, ao);

  gemm16<<<gemmBlocks, 256, panelBytes, stream>>>(ao, fo16, fo_b, d_out, M, 2);
}
